// CustomMultiheadAttention_21715354649879
// MI455X (gfx1250) — matrix operand order repaired and checked by reading, not compiled, not run on hardware
//
#include <hip/hip_runtime.h>
#include <cstdint>
#include <cstddef>

#define S_LEN  2048
#define E_DIM  1024
#define NHEAD  16
#define P_DIM  64
#define HDIM   64
#define SM_SCALE 0.125f   // 1/sqrt(64)

typedef __bf16 bf16_t;
typedef __attribute__((ext_vector_type(4)))  __bf16 v4bf;
typedef __attribute__((ext_vector_type(8)))  __bf16 v8bf;
typedef __attribute__((ext_vector_type(16))) __bf16 v16bf;
typedef __attribute__((ext_vector_type(4)))  float  v4f;
typedef __attribute__((ext_vector_type(8)))  float  v8f;

// ---------------------------------------------------------------------------
// WMMA helpers (CDNA5: wave32, D = A(16x32 bf16) * B(32x16 bf16) + C(16x16 f32))
// ---------------------------------------------------------------------------
__device__ __forceinline__ v8f wmma_bf16(v16bf a, v16bf b, v8f c) {
  // (neg_a, A, neg_b, B, c_mod, C, reuse_a, reuse_b)
  return __builtin_amdgcn_wmma_f32_16x16x32_bf16(false, a, false, b, (short)0, c,
                                                 false, false);
}

// A-matrix fragment, 16x32 bf16, source row-major [M][K] with row stride ld.
// Lane layout (ISA 7.12.2): lanes 0-15 hold K 0-7 & 16-23 of row M=lane,
// lanes 16-31 hold K 8-15 & 24-31 of row M=lane-16.
__device__ __forceinline__ v16bf load_a_frag(const bf16_t* __restrict__ base,
                                             int ld, int row0, int col0, int lane) {
  const int half = lane >> 4, r = lane & 15;
  const bf16_t* p = base + (size_t)(row0 + r) * ld + col0 + half * 8;
  v8bf lo = *(const v8bf*)(p);        // K: half*8 + 0..7
  v8bf hi = *(const v8bf*)(p + 16);   // K: 16 + half*8 + 0..7
  v16bf a;
#pragma unroll
  for (int i = 0; i < 8; ++i) { a[i] = lo[i]; a[i + 8] = hi[i]; }
  return a;
}

// B-matrix fragment, 32x16 bf16 (KxN), built from row-major srcN[N][K]
// (i.e. computing X * srcN^T). Lanes 0-15: N=lane, K 0-7 & 16-23;
// lanes 16-31: N=lane-16, K 8-15 & 24-31.
__device__ __forceinline__ v16bf load_b_frag(const bf16_t* __restrict__ baseN,
                                             int ld, int n0, int k0, int lane) {
  const int half = lane >> 4, r = lane & 15;
  const bf16_t* p = baseN + (size_t)(n0 + r) * ld + k0 + half * 8;
  v8bf lo = *(const v8bf*)(p);
  v8bf hi = *(const v8bf*)(p + 16);
  v16bf b;
#pragma unroll
  for (int i = 0; i < 8; ++i) { b[i] = lo[i]; b[i + 8] = hi[i]; }
  return b;
}

// ---------------------------------------------------------------------------
// f32 -> bf16 bulk convert (vectorized: b128 load, b64 store)
// ---------------------------------------------------------------------------
__global__ void cvt_f32_to_bf16(const float* __restrict__ src,
                                bf16_t* __restrict__ dst, int n) {
  int i = (blockIdx.x * blockDim.x + threadIdx.x) * 4;
  if (i + 3 < n) {
    float4 v = *(const float4*)(src + i);
    v4bf o;
    o[0] = (bf16_t)v.x; o[1] = (bf16_t)v.y; o[2] = (bf16_t)v.z; o[3] = (bf16_t)v.w;
    *(v4bf*)(dst + i) = o;
  }
}

// ---------------------------------------------------------------------------
// GEMM: Y = A[MxK] * B[NxK]^T + bias[N]
//   MODE 0: Y bf16, row-major [M][N]
//   MODE 1: Y bf16, TRANSPOSED [N][M]   (used for V so PV B-frags are contiguous)
//   MODE 2: Y f32,  row-major [M][N]
// One wave owns a 16x64 output tile (4 accumulators, A fragment reused 4x).
// ---------------------------------------------------------------------------
template <int MODE>
__global__ void __launch_bounds__(256)
gemm_bf16_wmma(const bf16_t* __restrict__ A, const bf16_t* __restrict__ B,
               const float* __restrict__ bias, void* __restrict__ out,
               int M, int N, int K) {
  const int wave = blockIdx.x * (blockDim.x >> 5) + (threadIdx.x >> 5);
  const int lane = threadIdx.x & 31;
  const int ntn  = N >> 6;                  // N / 64
  const int m0   = (wave / ntn) * 16;
  const int n0   = (wave % ntn) * 64;
  if (m0 >= M) return;

  v8f acc[4] = {};
  for (int k0 = 0; k0 < K; k0 += 32) {
    v16bf a = load_a_frag(A, K, m0, k0, lane);
#pragma unroll
    for (int g = 0; g < 4; ++g) {
      v16bf b = load_b_frag(B, K, n0 + g * 16, k0, lane);
      acc[g] = wmma_bf16(a, b, acc[g]);
    }
  }

  const int half = lane >> 4, r = lane & 15;
#pragma unroll
  for (int g = 0; g < 4; ++g) {
    const int n  = n0 + g * 16 + r;
    const float bb = bias[n];
    if constexpr (MODE == 2) {
      float* O = (float*)out;
#pragma unroll
      for (int i = 0; i < 8; ++i) {
        const int m = m0 + half * 8 + i;     // C layout: vgpr i -> M = i (+8)
        O[(size_t)m * N + n] = acc[g][i] + bb;
      }
    } else if constexpr (MODE == 0) {
      bf16_t* O = (bf16_t*)out;
#pragma unroll
      for (int i = 0; i < 8; ++i) {
        const int m = m0 + half * 8 + i;
        O[(size_t)m * N + n] = (bf16_t)(acc[g][i] + bb);
      }
    } else {  // MODE 1: transposed, contiguous 16B vector store per group
      bf16_t* O = (bf16_t*)out;
      v8bf v;
#pragma unroll
      for (int i = 0; i < 8; ++i) v[i] = (bf16_t)(acc[g][i] + bb);
      *(v8bf*)(O + (size_t)n * M + m0 + half * 8) = v;
    }
  }
}

// ---------------------------------------------------------------------------
// Attention: one workgroup = (head h, 16-row block). 8 waves, 256 threads.
// LDS: 16x2048 f32 score strip (128 KB, lives in CDNA5's 320KB LDS) +
//      rowmax/rowinv + 8x16x64 f32 PV partials (32 KB).
// Phase 1: scores = q*k^T + sim*sim^T (4 WMMAs per 16x16 tile), *SCALE -> LDS
// Phase 2: per-row max & 1/sumexp via wave shuffles
// Phase 3: p = exp(s-max)/sum -> b128 stores of w to HBM + bf16 A-frags feeding
//          PV WMMAs against transposed V; cross-wave reduce -> attn (bf16)
// ---------------------------------------------------------------------------
__global__ void __launch_bounds__(256)
attention_wmma(const bf16_t* __restrict__ qb, const bf16_t* __restrict__ kb,
               const bf16_t* __restrict__ vT, const bf16_t* __restrict__ simb,
               float* __restrict__ w_out, bf16_t* __restrict__ attn_out) {
  extern __shared__ char smem_raw[];
  float* scores  = (float*)smem_raw;                 // 16 * 2048
  float* rowmax  = scores + 16 * S_LEN;              // 16
  float* rowinv  = rowmax + 16;                      // 16
  float* partial = rowinv + 16;                      // 8 * 16 * 64

  const int h    = blockIdx.y;
  const int s0   = blockIdx.x * 16;
  const int wave = threadIdx.x >> 5;
  const int lane = threadIdx.x & 31;
  const int half = lane >> 4, r = lane & 15;

  // A fragments for this row block (held in registers the whole kernel)
  const v16bf aq0 = load_a_frag(qb,   E_DIM, s0, h * HDIM + 0,  lane);
  const v16bf aq1 = load_a_frag(qb,   E_DIM, s0, h * HDIM + 32, lane);
  const v16bf as0 = load_a_frag(simb, P_DIM, s0, 0,  lane);
  const v16bf as1 = load_a_frag(simb, P_DIM, s0, 32, lane);

  // ---- Phase 1: score strip into LDS --------------------------------------
  for (int j = wave; j < S_LEN / 16; j += 8) {
    const int t0 = j * 16;
    v8f acc = {};
    acc = wmma_bf16(aq0, load_b_frag(kb,   E_DIM, t0, h * HDIM + 0,  lane), acc);
    acc = wmma_bf16(aq1, load_b_frag(kb,   E_DIM, t0, h * HDIM + 32, lane), acc);
    acc = wmma_bf16(as0, load_b_frag(simb, P_DIM, t0, 0,  lane), acc);
    acc = wmma_bf16(as1, load_b_frag(simb, P_DIM, t0, 32, lane), acc);
#pragma unroll
    for (int i = 0; i < 8; ++i) {
      const int m = half * 8 + i;
      scores[m * S_LEN + t0 + r] = acc[i] * SM_SCALE;
    }
  }
  __syncthreads();

  // ---- Phase 2: row max & inverse sumexp (2 rows per wave) ----------------
  for (int row = wave * 2; row < wave * 2 + 2; ++row) {
    const float* sr = scores + row * S_LEN;
    float mx = -1e30f;
    for (int c = lane; c < S_LEN; c += 32) mx = fmaxf(mx, sr[c]);
#pragma unroll
    for (int o = 16; o > 0; o >>= 1) mx = fmaxf(mx, __shfl_xor(mx, o, 32));
    float sum = 0.f;
    for (int c = lane; c < S_LEN; c += 32) sum += __expf(sr[c] - mx);
#pragma unroll
    for (int o = 16; o > 0; o >>= 1) sum += __shfl_xor(sum, o, 32);
    if (lane == 0) { rowmax[row] = mx; rowinv[row] = 1.f / sum; }
  }
  __syncthreads();

  // ---- Phase 3: normalize, stream w, PV WMMAs -----------------------------
  v8f acc[4] = {};
  const float mrow = rowmax[r];
  const float irow = rowinv[r];
  float* wrow = w_out + ((size_t)h * S_LEN + (s0 + r)) * S_LEN;
  const float* srow = scores + r * S_LEN;

  for (int step = 0; step < 8; ++step) {
    const int t0 = wave * 256 + step * 32;   // 32 K-columns per WMMA step
    const int c1 = t0 + half * 8;
    const int c2 = t0 + 16 + half * 8;
    v4f s1a = *(const v4f*)(srow + c1);
    v4f s1b = *(const v4f*)(srow + c1 + 4);
    v4f s2a = *(const v4f*)(srow + c2);
    v4f s2b = *(const v4f*)(srow + c2 + 4);
    float p1[8], p2[8];
#pragma unroll
    for (int i = 0; i < 4; ++i) {
      p1[i]     = __expf(s1a[i] - mrow) * irow;
      p1[i + 4] = __expf(s1b[i] - mrow) * irow;
      p2[i]     = __expf(s2a[i] - mrow) * irow;
      p2[i + 4] = __expf(s2b[i] - mrow) * irow;
    }
    // the single mandatory HBM write of w: 4x global_store_b128 per lane
    v4f w0, w1, w2, w3;
#pragma unroll
    for (int i = 0; i < 4; ++i) {
      w0[i] = p1[i]; w1[i] = p1[i + 4]; w2[i] = p2[i]; w3[i] = p2[i + 4];
    }
    *(v4f*)(wrow + c1)     = w0;
    *(v4f*)(wrow + c1 + 4) = w1;
    *(v4f*)(wrow + c2)     = w2;
    *(v4f*)(wrow + c2 + 4) = w3;
    // same values become the bf16 A fragment for P*V
    v16bf a;
#pragma unroll
    for (int i = 0; i < 8; ++i) { a[i] = (bf16_t)p1[i]; a[i + 8] = (bf16_t)p2[i]; }
#pragma unroll
    for (int g = 0; g < 4; ++g) {
      v16bf b = load_b_frag(vT, S_LEN, h * HDIM + g * 16, t0, lane);
      acc[g] = wmma_bf16(a, b, acc[g]);
    }
  }

  // stash this wave's 16x64 partial, then reduce across the 8 waves
  float* pp = partial + wave * (16 * 64);
#pragma unroll
  for (int g = 0; g < 4; ++g)
#pragma unroll
    for (int i = 0; i < 8; ++i)
      pp[(half * 8 + i) * 64 + g * 16 + r] = acc[g][i];
  __syncthreads();

  for (int e = threadIdx.x; e < 16 * 64; e += 256) {
    float s = 0.f;
#pragma unroll
    for (int w = 0; w < 8; ++w) s += partial[w * 1024 + e];
    const int m = e >> 6, d = e & 63;
    attn_out[(size_t)(s0 + m) * E_DIM + h * HDIM + d] = (bf16_t)s;
  }
}

// ---------------------------------------------------------------------------
// Host orchestration
// ---------------------------------------------------------------------------
extern "C" void kernel_launch(void* const* d_in, const int* in_sizes, int n_in,
                              void* d_out, int out_size, void* d_ws, size_t ws_size,
                              hipStream_t stream) {
  const float* query = (const float*)d_in[0];
  const float* key   = (const float*)d_in[1];
  const float* value = (const float*)d_in[2];
  const float* sim   = (const float*)d_in[3];
  const float* Wq = (const float*)d_in[4];  const float* bq = (const float*)d_in[5];
  const float* Wk = (const float*)d_in[6];  const float* bk = (const float*)d_in[7];
  const float* Wv = (const float*)d_in[8];  const float* bv = (const float*)d_in[9];
  const float* Wo = (const float*)d_in[10]; const float* bo = (const float*)d_in[11];

  float* out_p = (float*)d_out;                               // [S, E]
  float* w_p   = out_p + (size_t)S_LEN * E_DIM;               // [H, S, S]

  // carve bf16 workspace
  size_t off = 0;
  auto carve = [&](size_t nelem) {
    bf16_t* p = (bf16_t*)((char*)d_ws + off);
    off += ((nelem * sizeof(bf16_t)) + 255) & ~(size_t)255;
    return p;
  };
  const size_t SE = (size_t)S_LEN * E_DIM, EE = (size_t)E_DIM * E_DIM;
  bf16_t* query_bf = carve(SE);
  bf16_t* key_bf   = carve(SE);
  bf16_t* value_bf = carve(SE);
  bf16_t* sim_bf   = carve((size_t)S_LEN * P_DIM);
  bf16_t* Wq_bf    = carve(EE);
  bf16_t* Wk_bf    = carve(EE);
  bf16_t* Wv_bf    = carve(EE);
  bf16_t* Wo_bf    = carve(EE);
  bf16_t* q_bf     = carve(SE);            // [S, E] row-major
  bf16_t* k_bf     = carve(SE);            // [S, E] row-major
  bf16_t* vT_bf    = carve(SE);            // [E, S] transposed
  bf16_t* attn_bf  = carve(SE);            // [S, E] row-major
  (void)ws_size; (void)n_in; (void)in_sizes; (void)out_size;

  auto cvt = [&](const float* s, bf16_t* d, int n) {
    cvt_f32_to_bf16<<<(n / 4 + 255) / 256, 256, 0, stream>>>(s, d, n);
  };
  cvt(query, query_bf, SE);
  cvt(key,   key_bf,   SE);
  cvt(value, value_bf, SE);
  cvt(sim,   sim_bf,   S_LEN * P_DIM);
  cvt(Wq, Wq_bf, EE);
  cvt(Wk, Wk_bf, EE);
  cvt(Wv, Wv_bf, EE);
  cvt(Wo, Wo_bf, EE);

  // projections: (S/16)*(E/64) = 2048 waves -> 256 blocks of 8 waves
  const int gemm_blocks = (S_LEN / 16) * (E_DIM / 64) / 8;
  gemm_bf16_wmma<0><<<gemm_blocks, 256, 0, stream>>>(query_bf, Wq_bf, bq, q_bf,
                                                     S_LEN, E_DIM, E_DIM);
  gemm_bf16_wmma<0><<<gemm_blocks, 256, 0, stream>>>(key_bf, Wk_bf, bk, k_bf,
                                                     S_LEN, E_DIM, E_DIM);
  gemm_bf16_wmma<1><<<gemm_blocks, 256, 0, stream>>>(value_bf, Wv_bf, bv, vT_bf,
                                                     S_LEN, E_DIM, E_DIM);

  // attention: grid (S/16 row-blocks, H heads); 164KB dynamic LDS per WG
  const size_t attn_lds = (16 * S_LEN + 32 + 8 * 16 * 64) * sizeof(float);
  dim3 agrid(S_LEN / 16, NHEAD);
  attention_wmma<<<agrid, 256, attn_lds, stream>>>(q_bf, k_bf, vT_bf, sim_bf,
                                                   w_p, attn_bf);

  // output projection -> f32
  gemm_bf16_wmma<2><<<gemm_blocks, 256, 0, stream>>>(attn_bf, Wo_bf, bo, out_p,
                                                     S_LEN, E_DIM, E_DIM);
}
